// TaskHead_21371757265471
// MI455X (gfx1250) — compile-verified, matrix operand order checked by reading, not run on hardware
//
#include <hip/hip_runtime.h>
#include <hip/hip_bf16.h>

#define BB 2048
#define TT 512
#define DD 32
#define KK 10
#define NEG_INF (-3.402823466e38f)

typedef __attribute__((ext_vector_type(16))) _Float16 v16h;
typedef __attribute__((ext_vector_type(8)))  float    v8f;

// ds_swizzle_b32 xor-butterfly add within 16-lane groups (and_mask=0x1f keeps
// group-of-32 mode; xor masks < 16 never cross the 16-lane half).
#define SWZ_ADD(v, m) \
    ((v) + __int_as_float(__builtin_amdgcn_ds_swizzle(__float_as_int(v), ((m) << 10) | 0x1f)))

// ---------------------------------------------------------------------------
// Kernel 1: emissions = LayerNorm_K( x @ W^T + b ) * gamma + beta
// One wave computes a 16-row x 16-col tile with a single v_wmma_f32_16x16x32_f16
// (cols 10..15 zero-padded). LN stats via ds_swizzle butterflies.
// ---------------------------------------------------------------------------
__global__ __launch_bounds__(256) void emissions_wmma(
    const float* __restrict__ x, const float* __restrict__ W,
    const float* __restrict__ bias, const float* __restrict__ gamma,
    const float* __restrict__ beta, float* __restrict__ em)
{
    const int lane = threadIdx.x & 31;
    const int wave = threadIdx.x >> 5;
    const long tile = (long)blockIdx.x * 8 + wave;   // 16-row tile of flattened (b,t)
    const long row0 = tile * 16;

    const int half = lane >> 4;       // which 16-lane half
    const int sub  = lane & 15;       // A: row-in-tile, B/D: column (class)
    const int c1 = 8 * half;          // K chunk 1 start (ISA 16-bit A layout)
    const int c2 = 16 + 8 * half;     // K chunk 2 start

    // A: 16 rows of x, depth 32 (f16)
    v16h a;
    {
        const float* xr = x + (row0 + sub) * DD;
        #pragma unroll
        for (int j = 0; j < 8; ++j) {
            a[j]     = (_Float16)xr[c1 + j];
            a[8 + j] = (_Float16)xr[c2 + j];
        }
    }
    // B: W^T (32 x 16); column n = weight row n for n<10, zero otherwise
    v16h bm;
    if (sub < KK) {
        const float* wr = W + sub * DD;
        #pragma unroll
        for (int j = 0; j < 8; ++j) {
            bm[j]     = (_Float16)wr[c1 + j];
            bm[8 + j] = (_Float16)wr[c2 + j];
        }
    } else {
        #pragma unroll
        for (int j = 0; j < 16; ++j) bm[j] = (_Float16)0.f;
    }

    v8f c = {};
    c = __builtin_amdgcn_wmma_f32_16x16x32_f16(
            /*neg_a=*/false, a, /*neg_b=*/false, bm,
            /*c_mod=*/(short)0, c, /*reuse_a=*/false, /*reuse_b=*/false);

    const bool  valid = (sub < KK);
    const float bv = valid ? bias[sub]  : 0.f;
    const float gv = valid ? gamma[sub] : 0.f;
    const float be = valid ? beta[sub]  : 0.f;

    #pragma unroll
    for (int j = 0; j < 8; ++j) {
        // c[j]: row = j + 8*half (within tile), col = sub
        float h = valid ? (c[j] + bv) : 0.f;
        // sum over the 10 valid columns (invalid lanes contribute 0)
        float s = h;
        s = SWZ_ADD(s, 1); s = SWZ_ADD(s, 2);
        s = SWZ_ADD(s, 4); s = SWZ_ADD(s, 8);
        const float mu = s * (1.f / KK);
        const float d  = h - mu;
        float e = valid ? d * d : 0.f;
        e = SWZ_ADD(e, 1); e = SWZ_ADD(e, 2);
        e = SWZ_ADD(e, 4); e = SWZ_ADD(e, 8);
        const float inv = rsqrtf(e * (1.f / KK) + 1e-5f);
        if (valid) {
            const long row = row0 + j + 8 * half;
            em[row * KK + sub] = d * inv * gv + be;
        }
    }
}

// ---------------------------------------------------------------------------
// Kernel 2: CRF log-likelihood per batch (numerator + log Z), 1 thread/seq.
// ---------------------------------------------------------------------------
__global__ __launch_bounds__(256) void crf_forward(
    const float* __restrict__ em, const int* __restrict__ labels,
    const int* __restrict__ mask, const float* __restrict__ start_t,
    const float* __restrict__ end_t, const float* __restrict__ trans,
    float* __restrict__ llh)
{
    __shared__ float s_tr[KK * KK];
    __shared__ float s_st[KK], s_et[KK];
    for (int i = threadIdx.x; i < KK * KK; i += blockDim.x) s_tr[i] = trans[i];
    if (threadIdx.x < KK) {
        s_st[threadIdx.x] = start_t[threadIdx.x];
        s_et[threadIdx.x] = end_t[threadIdx.x];
    }
    __syncthreads();

    const int b = blockIdx.x * blockDim.x + threadIdx.x;
    if (b >= BB) return;
    const float* e  = em     + (long)b * TT * KK;
    const int* lab  = labels + (long)b * TT;
    const int* msk  = mask   + (long)b * TT;

    float score[KK];
    #pragma unroll
    for (int k = 0; k < KK; ++k) score[k] = s_st[k] + e[k];

    int   prev = lab[0];
    float num  = s_st[prev] + e[prev];
    int   cnt  = (msk[0] != 0);

    for (int t = 1; t < TT; ++t) {
        // L2-resident stream: prefetch ~8 steps ahead (global_prefetch_b8)
        if (t + 8 < TT) __builtin_prefetch(e + (t + 8) * KK, 0, 0);

        float et[KK];
        #pragma unroll
        for (int k = 0; k < KK; ++k) et[k] = e[t * KK + k];

        const int m   = msk[t];
        const int tag = lab[t];
        if (m) num += s_tr[prev * KK + tag] + e[t * KK + tag];
        prev = tag;
        cnt += (m != 0);

        float nxt[KK];
        #pragma unroll
        for (int j = 0; j < KK; ++j) {
            float mx = NEG_INF;
            #pragma unroll
            for (int i = 0; i < KK; ++i)
                mx = fmaxf(mx, score[i] + s_tr[i * KK + j]);
            float ss = 0.f;
            #pragma unroll
            for (int i = 0; i < KK; ++i)
                ss += __expf(score[i] + s_tr[i * KK + j] - mx);
            nxt[j] = mx + __logf(ss) + et[j];
        }
        if (m) {
            #pragma unroll
            for (int j = 0; j < KK; ++j) score[j] = nxt[j];
        }
    }

    float mz = NEG_INF;
    #pragma unroll
    for (int k = 0; k < KK; ++k) mz = fmaxf(mz, score[k] + s_et[k]);
    float ss = 0.f;
    #pragma unroll
    for (int k = 0; k < KK; ++k) ss += __expf(score[k] + s_et[k] - mz);
    const float logz = mz + __logf(ss);

    const int last_idx = (cnt > 0) ? (cnt - 1) : 0;
    num += s_et[lab[last_idx]];
    llh[b] = num - logz;
}

// ---------------------------------------------------------------------------
// Kernel 3: deterministic loss reduction: out[0] = -sum(llh)
// ---------------------------------------------------------------------------
__global__ __launch_bounds__(256) void reduce_loss(
    const float* __restrict__ llh, float* __restrict__ out)
{
    __shared__ float sm[256];
    float s = 0.f;
    for (int i = threadIdx.x; i < BB; i += 256) s += llh[i];
    sm[threadIdx.x] = s;
    __syncthreads();
    for (int off = 128; off > 0; off >>= 1) {
        if (threadIdx.x < off) sm[threadIdx.x] += sm[threadIdx.x + off];
        __syncthreads();
    }
    if (threadIdx.x == 0) out[0] = -sm[0];
}

// ---------------------------------------------------------------------------
// Kernel 4: Viterbi decode, 1 thread/seq. u8 backpointers in workspace,
// path written as float into d_out[1 + b*T + t]. First-max tie-break (strict >).
// ---------------------------------------------------------------------------
__global__ __launch_bounds__(256) void viterbi_decode(
    const float* __restrict__ em, const float* __restrict__ start_t,
    const float* __restrict__ end_t, const float* __restrict__ trans,
    unsigned char* __restrict__ hist, float* __restrict__ path)
{
    __shared__ float s_tr[KK * KK];
    __shared__ float s_st[KK], s_et[KK];
    for (int i = threadIdx.x; i < KK * KK; i += blockDim.x) s_tr[i] = trans[i];
    if (threadIdx.x < KK) {
        s_st[threadIdx.x] = start_t[threadIdx.x];
        s_et[threadIdx.x] = end_t[threadIdx.x];
    }
    __syncthreads();

    const int b = blockIdx.x * blockDim.x + threadIdx.x;
    if (b >= BB) return;
    const float* e = em + (long)b * TT * KK;
    unsigned char* hb = hist + (long)b * (TT - 1) * KK;

    float score[KK];
    #pragma unroll
    for (int k = 0; k < KK; ++k) score[k] = s_st[k] + e[k];

    for (int t = 1; t < TT; ++t) {
        if (t + 8 < TT) __builtin_prefetch(e + (t + 8) * KK, 0, 0);

        float nxt[KK];
        #pragma unroll
        for (int j = 0; j < KK; ++j) {
            float best = score[0] + s_tr[j];   // i = 0
            int   bi   = 0;
            #pragma unroll
            for (int i = 1; i < KK; ++i) {
                const float v = score[i] + s_tr[i * KK + j];
                if (v > best) { best = v; bi = i; }
            }
            nxt[j] = best + e[t * KK + j];
            hb[(long)(t - 1) * KK + j] = (unsigned char)bi;
        }
        #pragma unroll
        for (int j = 0; j < KK; ++j) score[j] = nxt[j];
    }

    float best = score[0] + s_et[0];
    int   cur  = 0;
    #pragma unroll
    for (int k = 1; k < KK; ++k) {
        const float v = score[k] + s_et[k];
        if (v > best) { best = v; cur = k; }
    }
    float* pb = path + (long)b * TT;
    pb[TT - 1] = (float)cur;
    for (int t = TT - 2; t >= 0; --t) {
        cur = hb[(long)t * KK + cur];
        pb[t] = (float)cur;
    }
}

// ---------------------------------------------------------------------------
extern "C" void kernel_launch(void* const* d_in, const int* in_sizes, int n_in,
                              void* d_out, int out_size, void* d_ws, size_t ws_size,
                              hipStream_t stream) {
    const float* x      = (const float*)d_in[0];
    const int*   labels = (const int*)  d_in[1];
    const int*   amask  = (const int*)  d_in[2];
    const float* W      = (const float*)d_in[3];
    const float* bias   = (const float*)d_in[4];
    const float* gamma  = (const float*)d_in[5];
    const float* beta   = (const float*)d_in[6];
    const float* st     = (const float*)d_in[7];
    const float* et     = (const float*)d_in[8];
    const float* tr     = (const float*)d_in[9];
    float* out = (float*)d_out;

    char* ws = (char*)d_ws;
    const size_t em_bytes  = (size_t)BB * TT * KK * sizeof(float);   // ~41.9 MB
    const size_t llh_bytes = (size_t)BB * sizeof(float);
    float*         em   = (float*)ws;
    float*         llh  = (float*)(ws + em_bytes);
    unsigned char* hist = (unsigned char*)(ws + em_bytes + llh_bytes); // ~10.5 MB

    // 1) emissions via WMMA: (B*T/16) tiles, 8 waves/block
    const int tiles   = (BB * TT) / 16;      // 65536
    const int eblocks = tiles / 8;           // 8192
    emissions_wmma<<<eblocks, 256, 0, stream>>>(x, W, bias, gamma, beta, em);

    // 2) CRF forward scan (1 thread per sequence)
    crf_forward<<<BB / 256, 256, 0, stream>>>(em, labels, amask, st, et, tr, llh);

    // 3) deterministic loss reduction
    reduce_loss<<<1, 256, 0, stream>>>(llh, out);

    // 4) Viterbi decode + backtrace into d_out[1..]
    viterbi_decode<<<BB / 256, 256, 0, stream>>>(em, st, et, tr, hist, out + 1);
}
